// MolPredFragFPv8_53970559042041
// MI455X (gfx1250) — matrix operand order repaired and checked by reading, not compiled, not run on hardware
//
#include <hip/hip_runtime.h>
#include <hip/hip_bf16.h>

// ---------------------------------------------------------------------------
// AttentiveFP forward on gfx1250 (MI455X). All linear_bn layers except align are folded
// into (f16 weight, f32 bias) and run through v_wmma_f32_16x16x32_f16.
// ---------------------------------------------------------------------------

typedef __attribute__((ext_vector_type(16))) _Float16 v16h;
typedef __attribute__((ext_vector_type(8)))  _Float16 v8h;
typedef __attribute__((ext_vector_type(8)))  float    v8f;

#define DEV static __device__ __forceinline__

constexpr int B_ = 128, A_ = 128, K_ = 6, NBOND_ = 128, FPD = 256;
constexpr int BA  = B_ * A_;        // 16384 atom rows
constexpr int BAK = B_ * A_ * K_;   // 98304 neighbor rows
constexpr float EPS = 1e-6f;

// ---------------- workspace layout (constexpr byte offsets) ----------------
constexpr size_t alup(size_t x) { return (x + 255) & ~size_t(255); }

constexpr size_t OFF_ATOM0W = 0;                                       // f16 [256,64]
constexpr size_t OFF_ATOM1W = alup(OFF_ATOM0W + 256 * 64 * 2);         // f16 [256,256]
constexpr size_t OFF_BOND0W = alup(OFF_ATOM1W + 256 * 256 * 2);        // f16 [256,32]
constexpr size_t OFF_BOND1W = alup(OFF_BOND0W + 256 * 32 * 2);         // f16 [256,256]
constexpr size_t OFF_NEI0W  = alup(OFF_BOND1W + 256 * 256 * 2);        // f16 [256,768]
constexpr size_t OFF_NEI1W  = alup(OFF_NEI0W + 256 * 768 * 2);         // f16 [256,256]
constexpr size_t OFF_ATTW   = alup(OFF_NEI1W + 256 * 256 * 2);         // f16 3x[256,256]
constexpr size_t OFF_WIH    = alup(OFF_ATTW + 3ull * 256 * 256 * 2);   // f16 3x[768,256]
constexpr size_t OFF_WHH    = alup(OFF_WIH + 3ull * 768 * 256 * 2);    // f16 3x[768,256]
constexpr size_t OFF_BIAS   = alup(OFF_WHH + 3ull * 768 * 256 * 2);    // f32 9 slots x256
constexpr size_t OFF_ALIGNW = alup(OFF_BIAS + 9 * 256 * 4);            // f32 3x[512]
constexpr size_t OFF_ALIGNB = alup(OFF_ALIGNW + 3 * 512 * 4);          // f32 3x1 (padded)
constexpr size_t OFF_AIN16  = alup(OFF_ALIGNB + 256);                  // f16 [BA,64]
constexpr size_t OFF_BIN16  = alup(OFF_AIN16 + (size_t)BA * 64 * 2);   // f16 [BA,32]
constexpr size_t OFF_NEIIN  = alup(OFF_BIN16 + (size_t)BA * 32 * 2);   // f16 [BAK,768]
constexpr size_t OFF_T16    = alup(OFF_NEIIN + (size_t)BAK * 768 * 2); // f16 [BAK,256]
constexpr size_t OFF_TMPF   = alup(OFF_T16 + (size_t)BAK * 256 * 2);   // f32 [BAK,256]
constexpr size_t OFF_ATOMFP = alup(OFF_TMPF + (size_t)BAK * 256 * 4);  // f32 [BA,256]
constexpr size_t OFF_ATOM16 = alup(OFF_ATOMFP + (size_t)BA * 256 * 4); // f16 [BA,256]
constexpr size_t OFF_BONDFP = alup(OFF_ATOM16 + (size_t)BA * 256 * 2); // f32 [BA,256]
constexpr size_t OFF_NEIFP  = alup(OFF_BONDFP + (size_t)BA * 256 * 4); // f32 [BAK,256]
constexpr size_t OFF_NEI16  = alup(OFF_NEIFP + (size_t)BAK * 256 * 4); // f16 [BAK,256]
constexpr size_t OFF_SCORE  = alup(OFF_NEI16 + (size_t)BAK * 256 * 2); // f32 [BAK]
constexpr size_t OFF_SA     = alup(OFF_SCORE + (size_t)BAK * 4);       // f32 [BA]
constexpr size_t OFF_CTX16  = alup(OFF_SA + (size_t)BA * 4);           // f16 [BA,256]
constexpr size_t OFF_GI     = alup(OFF_CTX16 + (size_t)BA * 256 * 2);  // f32 [BA,768]
constexpr size_t OFF_GH     = alup(OFF_GI + (size_t)BA * 768 * 4);     // f32 [BA,768]

// ---------------- preprocessing kernels ----------------

// Fold eval-mode BN into linear: W'[o,i] = W[o,i]*g[o]/sqrt(v[o]+eps) (f16,
// zero-padded to kpad); b'[o] = (b[o]-m[o])*s + be[o].
__global__ void fold_linbn_kernel(const float* __restrict__ W, const float* __restrict__ b,
                                  const float* __restrict__ be, const float* __restrict__ g,
                                  const float* __restrict__ m, const float* __restrict__ v,
                                  _Float16* __restrict__ Wout, float* __restrict__ bout,
                                  int dout, int din, int kpad) {
  int t = blockIdx.x * blockDim.x + threadIdx.x;
  if (t >= dout * kpad) return;
  int o = t / kpad, i = t - o * kpad;
  float s = g[o] * rsqrtf(v[o] + EPS);
  Wout[t] = (i < din) ? (_Float16)(W[(size_t)o * din + i] * s) : (_Float16)0.f;
  if (i == 0) bout[o] = (b[o] - m[o]) * s + be[o];
}

// align linear_bn has dout==1: keep folded weight in f32.
__global__ void fold_align_kernel(const float* __restrict__ W, const float* __restrict__ b,
                                  const float* __restrict__ be, const float* __restrict__ g,
                                  const float* __restrict__ m, const float* __restrict__ v,
                                  float* __restrict__ wout, float* __restrict__ bout) {
  int t = blockIdx.x * blockDim.x + threadIdx.x;
  if (t >= 512) return;
  float s = g[0] * rsqrtf(v[0] + EPS);
  wout[t] = W[t] * s;
  if (t == 0) bout[0] = (b[0] - m[0]) * s + be[0];
}

__global__ void cvt16_kernel(const float* __restrict__ in, _Float16* __restrict__ out, int n) {
  int t = blockIdx.x * blockDim.x + threadIdx.x;
  if (t < n) out[t] = (_Float16)in[t];
}

// pad + convert input features to f16 rows of width kpad
__global__ void padcvt_kernel(const float* __restrict__ in, _Float16* __restrict__ out,
                              int rows, int din, int kpad) {
  int t = blockIdx.x * blockDim.x + threadIdx.x;
  if (t >= rows * kpad) return;
  int r = t / kpad, i = t - r * kpad;
  out[t] = (i < din) ? (_Float16)in[(size_t)r * din + i] : (_Float16)0.f;
}

// ---------------- WMMA GEMM: C = act(A[M,K] x W[N,K]^T + bias) ----------------
// Fragment layouts used here (wave32; contiguous k, not the ISA table order):
//   A 16x32 f16 : lane l holds row M=l&15; K chunk  {kb..kb+15},             kb=16*(l>>4)
//   B 32x16 f16 : lane l holds col N=l&15; K chunk  {kb..kb+15},             kb=16*(l>>4)
//   C/D 16x16 f32: vgpr r, lane l -> (M = r + 8*(l>>4), N = l&15)
DEV v16h cat16(v8h lo, v8h hi) {
  v16h r;
#pragma unroll
  for (int i = 0; i < 8; ++i) { r[i] = lo[i]; r[i + 8] = hi[i]; }
  return r;
}

__global__ __launch_bounds__(128)
void gemm_wmma_kernel(const _Float16* __restrict__ A, const _Float16* __restrict__ W,
                      const float* __restrict__ bias, float* __restrict__ C,
                      _Float16* __restrict__ C16, int M, int N, int Kd, int act) {
  const int lane = threadIdx.x & 31;
  const int wv = threadIdx.x >> 5;
  const int m0 = (blockIdx.x * 4 + wv) * 16;   // wave-uniform -> EXEC stays full
  const int n0 = blockIdx.y * 64;
  if (m0 >= M) return;

  v8f acc0 = {}, acc1 = {}, acc2 = {}, acc3 = {};
  const int ar = m0 + (lane & 15);
  const int akb = (lane >> 4) << 4;
  const int bkb = (lane >> 4) << 4;
  const _Float16* brow = W + (size_t)(n0 + (lane & 15)) * Kd + bkb;

  for (int k0 = 0; k0 < Kd; k0 += 32) {
    const _Float16* ap = A + (size_t)ar * Kd + k0 + akb;
    v16h af = cat16(*(const v8h*)ap, *(const v8h*)(ap + 8));
    const _Float16* bp = brow + k0;
    v16h b0 = cat16(*(const v8h*)bp, *(const v8h*)(bp + 8));
    bp += (size_t)16 * Kd;
    v16h b1 = cat16(*(const v8h*)bp, *(const v8h*)(bp + 8));
    bp += (size_t)16 * Kd;
    v16h b2 = cat16(*(const v8h*)bp, *(const v8h*)(bp + 8));
    bp += (size_t)16 * Kd;
    v16h b3 = cat16(*(const v8h*)bp, *(const v8h*)(bp + 8));
    acc0 = __builtin_amdgcn_wmma_f32_16x16x32_f16(false, af, false, b0, (short)0, acc0, false, false);
    acc1 = __builtin_amdgcn_wmma_f32_16x16x32_f16(false, af, false, b1, (short)0, acc1, false, false);
    acc2 = __builtin_amdgcn_wmma_f32_16x16x32_f16(false, af, false, b2, (short)0, acc2, false, false);
    acc3 = __builtin_amdgcn_wmma_f32_16x16x32_f16(false, af, false, b3, (short)0, acc3, false, false);
  }

  const int nl = lane & 15;
  const int mr = m0 + ((lane >> 4) << 3);
  v8f accs[4] = {acc0, acc1, acc2, acc3};
#pragma unroll
  for (int j = 0; j < 4; ++j) {
    const int n = n0 + 16 * j + nl;
    const float bb = bias[n];
#pragma unroll
    for (int r = 0; r < 8; ++r) {
      float y = accs[j][r] + bb;
      if (act) y = fmaxf(y, 0.f);
      const size_t idx = (size_t)(mr + r) * N + n;
      C[idx] = y;
      if (C16) C16[idx] = (_Float16)y;
    }
  }
}

// ---------------- pointwise / gather / attention kernels ----------------

// build nei_fc input: [a_nb | b_nb | a+b-a*b] in f16, row per (b,a,k)
__global__ void buildnei_kernel(const float* __restrict__ atomFP, const float* __restrict__ bondFP,
                                const int* __restrict__ anl, const int* __restrict__ bnl,
                                _Float16* __restrict__ nein) {
  int row = blockIdx.x, f = threadIdx.x;
  int b = row / (A_ * K_);
  float a = atomFP[((size_t)b * A_ + anl[row]) * FPD + f];
  float bb = bondFP[((size_t)b * NBOND_ + bnl[row]) * FPD + f];
  float mx = a + bb - a * bb;
  size_t o = (size_t)row * 768;
  nein[o + f] = (_Float16)a;
  nein[o + 256 + f] = (_Float16)bb;
  nein[o + 512 + f] = (_Float16)mx;
}

// wave-per-row dot(X[row,0:256], w) -> out[row]
__global__ void rowdot_kernel(const float* __restrict__ X, const float* __restrict__ w,
                              float* __restrict__ out, int rows) {
  int lane = threadIdx.x & 31, wv = threadIdx.x >> 5;
  int row = blockIdx.x * (blockDim.x >> 5) + wv;
  if (row >= rows) return;
  const float* x = X + (size_t)row * FPD;
  float s = 0.f;
  for (int i = lane; i < FPD; i += 32) s += x[i] * w[i];
#pragma unroll
  for (int o = 16; o; o >>= 1) s += __shfl_xor(s, o, 32);
  if (lane == 0) out[row] = s;
}

// score[row] = leaky_relu(dot(neiFP[row], wN) + sA[row/K] + b) + softmax_mask
__global__ void score_kernel(const float* __restrict__ neiFP, const float* __restrict__ wN,
                             const float* __restrict__ sA, const float* __restrict__ ab,
                             const int* __restrict__ anl, float* __restrict__ score, int rows) {
  int lane = threadIdx.x & 31, wv = threadIdx.x >> 5;
  int row = blockIdx.x * (blockDim.x >> 5) + wv;
  if (row >= rows) return;
  const float* x = neiFP + (size_t)row * FPD;
  float s = 0.f;
  for (int i = lane; i < FPD; i += 32) s += x[i] * wN[i];
#pragma unroll
  for (int o = 16; o; o >>= 1) s += __shfl_xor(s, o, 32);
  if (lane == 0) {
    float t = s + sA[row / K_] + ab[0];
    float lr = t > 0.f ? t : 0.01f * t;              // jax leaky_relu slope 0.01
    score[row] = lr + ((anl[row] == A_ - 1) ? -9e8f : 0.f);
  }
}

// per (b,a): masked softmax over K, ctx = elu(sum attn*attended) -> f16
__global__ void ctx_kernel(const float* __restrict__ score, const float* __restrict__ attended,
                           const int* __restrict__ anl, _Float16* __restrict__ ctx16) {
  int ba = blockIdx.x, f = threadIdx.x;
  float sc[K_], am[K_];
#pragma unroll
  for (int k = 0; k < K_; ++k) {
    sc[k] = score[ba * K_ + k];
    am[k] = (anl[ba * K_ + k] == A_ - 1) ? 0.f : 1.f;
  }
  float mx = sc[0];
#pragma unroll
  for (int k = 1; k < K_; ++k) mx = fmaxf(mx, sc[k]);
  float e[K_], sum = 0.f;
#pragma unroll
  for (int k = 0; k < K_; ++k) { e[k] = __expf(sc[k] - mx); sum += e[k]; }
  float inv = 1.f / sum;
  float acc = 0.f;
#pragma unroll
  for (int k = 0; k < K_; ++k)
    acc += (e[k] * inv) * am[k] * attended[((size_t)ba * K_ + k) * FPD + f];
  float c = acc > 0.f ? acc : expm1f(acc);           // elu(alpha=1)
  ctx16[(size_t)ba * FPD + f] = (_Float16)c;
}

// GRU cell pointwise from precomputed gi=x*Wih^T+bih, gh=h*Whh^T+bhh
__global__ void gru_kernel(const float* __restrict__ gi, const float* __restrict__ gh,
                           const float* __restrict__ h, float* __restrict__ outF,
                           _Float16* __restrict__ out16, int n) {
  int t = blockIdx.x * blockDim.x + threadIdx.x;
  if (t >= n) return;
  int row = t >> 8, f = t & 255;
  size_t g0 = (size_t)row * 768 + f;
  float ir = gi[g0], iz = gi[g0 + 256], in = gi[g0 + 512];
  float hr = gh[g0], hz = gh[g0 + 256], hn = gh[g0 + 512];
  float r = 1.f / (1.f + __expf(-(ir + hr)));
  float z = 1.f / (1.f + __expf(-(iz + hz)));
  float nn = tanhf(in + r * hn);
  float hv = h[t];
  float o = (1.f - z) * nn + z * hv;
  outF[t] = o;
  out16[t] = (_Float16)o;
}

// neighbor_FP = relu(atom_FP)[gather anl] -> f32 + f16 copies
__global__ void gather_relu_kernel(const float* __restrict__ atomFP, const int* __restrict__ anl,
                                   float* __restrict__ neiFP, _Float16* __restrict__ nei16) {
  int row = blockIdx.x, f = threadIdx.x;
  int b = row / (A_ * K_);
  float v = fmaxf(atomFP[((size_t)b * A_ + anl[row]) * FPD + f], 0.f);
  size_t idx = (size_t)row * FPD + f;
  neiFP[idx] = v;
  nei16[idx] = (_Float16)v;
}

// ---------------- host-side launcher ----------------
static inline dim3 g1(int n, int b) { return dim3((n + b - 1) / b); }

extern "C" void kernel_launch(void* const* d_in, const int* in_sizes, int n_in,
                              void* d_out, int out_size, void* d_ws, size_t ws_size,
                              hipStream_t stream) {
  (void)in_sizes; (void)n_in; (void)out_size; (void)ws_size;
  char* ws = (char*)d_ws;

  // --- input pointers (JAX pytree flatten: dicts in sorted-key order) ---
  // 0..3: atom_features f32[128,128,39], bond_features f32[128,128,10],
  //       atom_neighbor_list i32[128,128,6], bond_neighbor_list i32[128,128,6]
  // params keys sorted: atom_fc, bond_fc, layers, nei_fc
  //   _lin_bn leaves sorted: W, b, be, g, m, v
  //   layer leaves sorted: align(6), attend(6), gru{W_hh,W_ih,b_hh,b_ih}
  const float* atom_feat = (const float*)d_in[0];
  const float* bond_feat = (const float*)d_in[1];
  const int* anl = (const int*)d_in[2];
  const int* bnl = (const int*)d_in[3];
  auto F = [&](int i) { return (const float*)d_in[i]; };
  const int ATOM0 = 4, ATOM1 = 10, BOND0 = 16, BOND1 = 22, LAY0 = 28, NEI0 = 76, NEI1 = 82;

  // --- workspace views ---
  _Float16* atom0W = (_Float16*)(ws + OFF_ATOM0W);
  _Float16* atom1W = (_Float16*)(ws + OFF_ATOM1W);
  _Float16* bond0W = (_Float16*)(ws + OFF_BOND0W);
  _Float16* bond1W = (_Float16*)(ws + OFF_BOND1W);
  _Float16* nei0W  = (_Float16*)(ws + OFF_NEI0W);
  _Float16* nei1W  = (_Float16*)(ws + OFF_NEI1W);
  _Float16* attW   = (_Float16*)(ws + OFF_ATTW);   // 3 x 65536
  _Float16* wih    = (_Float16*)(ws + OFF_WIH);    // 3 x 196608
  _Float16* whh    = (_Float16*)(ws + OFF_WHH);
  float* biasB     = (float*)(ws + OFF_BIAS);      // slots of 256
  float* alignW    = (float*)(ws + OFF_ALIGNW);    // 3 x 512
  float* alignB    = (float*)(ws + OFF_ALIGNB);
  _Float16* ain16  = (_Float16*)(ws + OFF_AIN16);
  _Float16* bin16  = (_Float16*)(ws + OFF_BIN16);
  _Float16* neiin  = (_Float16*)(ws + OFF_NEIIN);
  _Float16* t16    = (_Float16*)(ws + OFF_T16);
  float* tmpF      = (float*)(ws + OFF_TMPF);
  float* atomFP    = (float*)(ws + OFF_ATOMFP);
  _Float16* atom16 = (_Float16*)(ws + OFF_ATOM16);
  float* bondFP    = (float*)(ws + OFF_BONDFP);
  float* neiFP     = (float*)(ws + OFF_NEIFP);
  _Float16* nei16  = (_Float16*)(ws + OFF_NEI16);
  float* score     = (float*)(ws + OFF_SCORE);
  float* sA        = (float*)(ws + OFF_SA);
  _Float16* ctx16  = (_Float16*)(ws + OFF_CTX16);
  float* gi        = (float*)(ws + OFF_GI);
  float* gh        = (float*)(ws + OFF_GH);
  float* outF      = (float*)d_out;

  // --- 1) fold BN into weights, convert GRU weights, pad inputs ---
  auto fold = [&](int base, _Float16* Wout, float* bout, int dout, int din, int kpad) {
    fold_linbn_kernel<<<g1(dout * kpad, 256), 256, 0, stream>>>(
        F(base + 0), F(base + 1), F(base + 2), F(base + 3), F(base + 4), F(base + 5),
        Wout, bout, dout, din, kpad);
  };
  fold(ATOM0, atom0W, biasB + 0 * 256, 256, 39, 64);
  fold(ATOM1, atom1W, biasB + 1 * 256, 256, 256, 256);
  fold(BOND0, bond0W, biasB + 2 * 256, 256, 10, 32);
  fold(BOND1, bond1W, biasB + 3 * 256, 256, 256, 256);
  fold(NEI0,  nei0W,  biasB + 4 * 256, 256, 768, 768);
  fold(NEI1,  nei1W,  biasB + 5 * 256, 256, 256, 256);
  for (int l = 0; l < 3; ++l) {
    int base = LAY0 + 16 * l;
    fold_align_kernel<<<g1(512, 256), 256, 0, stream>>>(
        F(base + 0), F(base + 1), F(base + 2), F(base + 3), F(base + 4), F(base + 5),
        alignW + 512 * l, alignB + l);
    fold(base + 6, attW + 65536 * l, biasB + (6 + l) * 256, 256, 256, 256);
    cvt16_kernel<<<g1(768 * 256, 256), 256, 0, stream>>>(F(base + 12), whh + 196608 * l, 768 * 256);
    cvt16_kernel<<<g1(768 * 256, 256), 256, 0, stream>>>(F(base + 13), wih + 196608 * l, 768 * 256);
  }
  padcvt_kernel<<<g1(BA * 64, 256), 256, 0, stream>>>(atom_feat, ain16, BA, 39, 64);
  padcvt_kernel<<<g1(BA * 32, 256), 256, 0, stream>>>(bond_feat, bin16, BA, 10, 32);

  // --- 2) FPInitializer: atom_fc / bond_fc / nei_fc (all relu epilogues) ---
  auto gemm = [&](const _Float16* A, const _Float16* W, const float* b, float* C,
                  _Float16* C16, int M, int N, int Kd, int act) {
    gemm_wmma_kernel<<<dim3(M / 64, N / 64), 128, 0, stream>>>(A, W, b, C, C16, M, N, Kd, act);
  };
  gemm(ain16, atom0W, biasB + 0 * 256, tmpF, t16, BA, 256, 64, 1);
  gemm(t16, atom1W, biasB + 1 * 256, atomFP, atom16, BA, 256, 256, 1);
  gemm(bin16, bond0W, biasB + 2 * 256, tmpF, t16, BA, 256, 32, 1);
  gemm(t16, bond1W, biasB + 3 * 256, bondFP, nullptr, BA, 256, 256, 1);
  buildnei_kernel<<<BAK, 256, 0, stream>>>(atomFP, bondFP, anl, bnl, neiin);
  gemm(neiin, nei0W, biasB + 4 * 256, tmpF, t16, BAK, 256, 768, 1);
  gemm(t16, nei1W, biasB + 5 * 256, neiFP, nei16, BAK, 256, 256, 1);

  // --- 3) AttentiveFP layers ---
  for (int l = 0; l < 3; ++l) {
    int base = LAY0 + 16 * l;
    const float* bhh = F(base + 14);
    const float* bih = F(base + 15);
    rowdot_kernel<<<g1(BA, 8) /*8 waves/blk*/, 256, 0, stream>>>(atomFP, alignW + 512 * l, sA, BA);
    score_kernel<<<g1(BAK, 8), 256, 0, stream>>>(neiFP, alignW + 512 * l + 256, sA,
                                                 alignB + l, anl, score, BAK);
    gemm(nei16, attW + 65536 * l, biasB + (6 + l) * 256, tmpF, nullptr, BAK, 256, 256, 0);
    ctx_kernel<<<BA, 256, 0, stream>>>(score, tmpF, anl, ctx16);
    gemm(ctx16, wih + 196608 * l, bih, gi, nullptr, BA, 768, 256, 0);
    gemm(atom16, whh + 196608 * l, bhh, gh, nullptr, BA, 768, 256, 0);
    float* dst = (l == 2) ? outF : atomFP;
    gru_kernel<<<g1(BA * 256, 256), 256, 0, stream>>>(gi, gh, atomFP, dst, atom16, BA * 256);
    if (l < 2)
      gather_relu_kernel<<<BAK, 256, 0, stream>>>(atomFP, anl, neiFP, nei16);
  }
}
